// GCN_44830868636165
// MI455X (gfx1250) — matrix operand order repaired and checked by reading, not compiled, not run on hardware
//
#include <hip/hip_runtime.h>
#include <hip/hip_bf16.h>

// ---------------------------------------------------------------------------
// GCN forward on MI455X (gfx1250), bf16 WMMA + f32 accum, double-buffered LDS,
// async global->LDS copies (ASYNCcnt) for bf16 tiles, load+cvt for f32 adj.
//   agg1 = adj@x ; h = relu([x,agg1]@W1+b1)
//   agg2 = adj@h ; out = relu([h,agg2]@W2+b2) + h
// ---------------------------------------------------------------------------

typedef __attribute__((ext_vector_type(16))) __bf16 v16bf;
typedef __attribute__((ext_vector_type(8)))  float  v8f;

#define NN     8192
#define DD     512
#define TILE_M 128
#define TILE_N 128
#define TILE_K 64
#define LDSTR  72   // LDS row stride in bf16 (64 + 8 pad) = 144B (16B multiple)

union FragU { v16bf v; uint4 q[2]; };

// 64B contiguous global -> LDS async copy (per-lane), 4x b128.
// INST_OFFSET applies to BOTH global and LDS addresses (ISA 15.18.3 / 8.4.4).
__device__ __forceinline__ void async_copy64(const __bf16* gsrc, __bf16* ldst) {
  unsigned lds = (unsigned)(size_t)(void*)ldst;  // low 32 bits = LDS offset
  asm volatile(
      "global_load_async_to_lds_b128 %0, %1, off\n\t"
      "global_load_async_to_lds_b128 %0, %1, off offset:16\n\t"
      "global_load_async_to_lds_b128 %0, %1, off offset:32\n\t"
      "global_load_async_to_lds_b128 %0, %1, off offset:48"
      :: "v"(lds), "v"(gsrc) : "memory");
}
__device__ __forceinline__ void async_wait0() {
  asm volatile("s_wait_asynccnt 0x0" ::: "memory");
}

// ---------------------------------------------------------------------------
// prep kernels: f32 -> bf16 conversions / transposes
// ---------------------------------------------------------------------------
__global__ __launch_bounds__(256) void prep_x(const float* __restrict__ x,
                                              __bf16* __restrict__ xT,
                                              __bf16* __restrict__ cat1) {
  int i = blockIdx.x * 256 + threadIdx.x;      // over NN*DD
  int n = i >> 9, f = i & 511;
  __bf16 b = (__bf16)x[i];
  xT[(size_t)f * NN + n]     = b;              // [feat][node]: SpMM B operand
  cat1[(size_t)n * 1024 + f] = b;              // first half of cat1
}

__global__ __launch_bounds__(256) void prep_w(const float* __restrict__ W,
                                              __bf16* __restrict__ Wt) {
  int i = blockIdx.x * 256 + threadIdx.x;      // over 1024*512, W [in][out]
  int in = i >> 9, o = i & 511;
  Wt[(size_t)o * 1024 + in] = (__bf16)W[i];    // Wt [out][in]
}

// ---------------------------------------------------------------------------
// C[M,Nn] = A[M,K] * Bt[Nn,K]^T ; 128x128 tile, BK=64, 8 waves (2x4),
// 16 WMMAs / wave / K-step, ping-pong LDS buffers.
// ---------------------------------------------------------------------------
template <bool A_F32, int EPI>
__global__ __launch_bounds__(256)
void gemm_bf16_wmma(const void* __restrict__ Av,
                    const __bf16* __restrict__ Bt,
                    int K,
                    const float* __restrict__ bias,
                    __bf16* __restrict__ catOut, int catLd, int catOff,
                    float*  __restrict__ outF32, int outLd,
                    __bf16* __restrict__ outT,   int outTLd,
                    const float* __restrict__ resid) {
  __shared__ alignas(16) __bf16 As[2][TILE_M * LDSTR];
  __shared__ alignas(16) __bf16 Bs[2][TILE_N * LDSTR];

  const int t     = threadIdx.x;
  const int lane  = t & 31;
  const int wave  = t >> 5;
  const int waveM = wave & 1;    // 2 waves along M
  const int waveN = wave >> 1;   // 4 waves along N
  const int mBlock = blockIdx.y * TILE_M;
  const int nBlock = blockIdx.x * TILE_N;

  const int sRow  = t >> 1;      // staging: 2 threads/row, 32 elems each
  const int sSeg  = t & 1;
  const int lhalf = lane >> 4;
  const int l16   = lane & 15;

  const float*  Af = (const float*)Av;
  const __bf16* Ab = (const __bf16*)Av;

  v8f zero;
#pragma unroll
  for (int e = 0; e < 8; ++e) zero[e] = 0.f;
  v8f acc[4][2];
#pragma unroll
  for (int i = 0; i < 4; ++i)
#pragma unroll
    for (int j = 0; j < 2; ++j) acc[i][j] = zero;

  auto stageA = [&](int buf, int k0) {
    size_t gOff = (size_t)(mBlock + sRow) * K + k0 + sSeg * 32;
    __bf16* dst = &As[buf][sRow * LDSTR + sSeg * 32];
    if (A_F32) {
      const float4* p4 = (const float4*)(Af + gOff);
      alignas(16) __bf16 tmp[32];
#pragma unroll
      for (int q = 0; q < 8; ++q) {
        float4 f = p4[q];
        tmp[q * 4 + 0] = (__bf16)f.x; tmp[q * 4 + 1] = (__bf16)f.y;
        tmp[q * 4 + 2] = (__bf16)f.z; tmp[q * 4 + 3] = (__bf16)f.w;
      }
#pragma unroll
      for (int q = 0; q < 4; ++q) ((uint4*)dst)[q] = ((const uint4*)tmp)[q];
      if (k0 + TILE_K < K) __builtin_prefetch(Af + gOff + TILE_K, 0, 3);
    } else {
      async_copy64(Ab + gOff, dst);
    }
  };
  auto stageB = [&](int buf, int k0) {
    size_t gOff = (size_t)(nBlock + sRow) * K + k0 + sSeg * 32;
    async_copy64(Bt + gOff, &Bs[buf][sRow * LDSTR + sSeg * 32]);
  };

  // prologue: fill buffer 0
  stageA(0, 0);
  stageB(0, 0);
  async_wait0();
  __syncthreads();

  int p = 0;
  for (int k0 = 0; k0 < K; k0 += TILE_K) {
    const int kn = k0 + TILE_K;
    if (kn < K) { stageA(1 - p, kn); stageB(1 - p, kn); }  // fill 1-p async

    // consume buffer p: two k-steps of 32
#pragma unroll
    for (int kk = 0; kk < TILE_K; kk += 32) {
      FragU fa[4], fb[2];
#pragma unroll
      for (int jn = 0; jn < 2; ++jn) {
        // B frag: lane = col n=l16, K in [kk+lhalf*16, +16) contiguous
        const __bf16* bp =
            &Bs[p][(waveN * 32 + jn * 16 + l16) * LDSTR + kk + lhalf * 16];
        fb[jn].q[0] = *(const uint4*)(bp);
        fb[jn].q[1] = *(const uint4*)(bp + 8);
      }
#pragma unroll
      for (int im = 0; im < 4; ++im) {
        // A frag: lane = row m=l16, K in [kk+lhalf*16, +16) contiguous
        const __bf16* ap =
            &As[p][(waveM * 64 + im * 16 + l16) * LDSTR + kk + lhalf * 16];
        fa[im].q[0] = *(const uint4*)(ap);
        fa[im].q[1] = *(const uint4*)(ap + 8);
      }
#pragma unroll
      for (int im = 0; im < 4; ++im)
#pragma unroll
        for (int jn = 0; jn < 2; ++jn)
          acc[im][jn] = __builtin_amdgcn_wmma_f32_16x16x32_bf16(
              false, fa[im].v, false, fb[jn].v, (short)0, acc[im][jn],
              false, false);
    }

    async_wait0();      // this wave's async fills of 1-p are done
    __syncthreads();    // publish 1-p, release p
    p ^= 1;
  }

  // ---- epilogue: C layout m = r + 8*lhalf, n = l16 (per 16x16 tile) ----
#pragma unroll
  for (int im = 0; im < 4; ++im) {
#pragma unroll
    for (int jn = 0; jn < 2; ++jn) {
      const int gn = nBlock + waveN * 32 + jn * 16 + l16;
      float b = 0.f;
      if (EPI != 0) b = bias[gn];
#pragma unroll
      for (int r = 0; r < 8; ++r) {
        const int gm = mBlock + waveM * 64 + im * 16 + lhalf * 8 + r;
        float v = acc[im][jn][r];
        if (EPI == 0) {
          catOut[(size_t)gm * catLd + catOff + gn] = (__bf16)v;
        } else if (EPI == 1) {
          v += b; v = v > 0.f ? v : 0.f;
          outF32[(size_t)gm * outLd + gn] = v;          // h (f32 residual)
          outT[(size_t)gn * outTLd + gm]  = (__bf16)v;  // hT (SpMM2 B)
          catOut[(size_t)gm * catLd + gn] = (__bf16)v;  // cat2[:, :512]
        } else {
          v += b; v = v > 0.f ? v : 0.f;                // relu in graph_conv
          v += resid[(size_t)gm * outLd + gn];          // + h (h>=0)
          v = v > 0.f ? v : 0.f;
          outF32[(size_t)gm * outLd + gn] = v;          // d_out
        }
      }
    }
  }
}

// ---------------------------------------------------------------------------
extern "C" void kernel_launch(void* const* d_in, const int* in_sizes, int n_in,
                              void* d_out, int out_size, void* d_ws, size_t ws_size,
                              hipStream_t stream) {
  const float* x   = (const float*)d_in[0];
  const float* adj = (const float*)d_in[1];
  const float* W1  = (const float*)d_in[2];
  const float* b1  = (const float*)d_in[3];
  const float* W2  = (const float*)d_in[4];
  const float* b2  = (const float*)d_in[5];
  float* out = (float*)d_out;

  char* ws = (char*)d_ws;
  __bf16* xT   = (__bf16*)(ws);                        //  8 MB  [512][8192]
  __bf16* cat1 = (__bf16*)(ws + ( 8ull << 20));        // 16 MB  [8192][1024]
  __bf16* W1T  = (__bf16*)(ws + (24ull << 20));        //  1 MB  [512][1024]
  __bf16* W2T  = (__bf16*)(ws + (25ull << 20));        //  1 MB
  float*  hF   = (float*) (ws + (26ull << 20));        // 16 MB  [8192][512]
  __bf16* hT   = (__bf16*)(ws + (42ull << 20));        //  8 MB  [512][8192]
  __bf16* cat2 = (__bf16*)(ws + (50ull << 20));        // 16 MB  -> 66 MB total

  prep_x<<<(NN * DD) / 256, 256, 0, stream>>>(x, xT, cat1);
  prep_w<<<(1024 * 512) / 256, 256, 0, stream>>>(W1, W1T);
  prep_w<<<(1024 * 512) / 256, 256, 0, stream>>>(W2, W2T);

  dim3 g(DD / TILE_N, NN / TILE_M);  // (4, 64)

  // SpMM1: cat1[:,512:] = adj @ x
  gemm_bf16_wmma<true, 0><<<g, 256, 0, stream>>>(
      adj, xT, NN, nullptr, cat1, 1024, 512, nullptr, 0, nullptr, 0, nullptr);
  // dense1: h = relu(cat1 @ W1 + b1) -> hF, hT, cat2[:, :512]
  gemm_bf16_wmma<false, 1><<<g, 256, 0, stream>>>(
      cat1, W1T, 1024, b1, cat2, 1024, 0, hF, 512, hT, NN, nullptr);
  // SpMM2: cat2[:,512:] = adj @ h
  gemm_bf16_wmma<true, 0><<<g, 256, 0, stream>>>(
      adj, hT, NN, nullptr, cat2, 1024, 512, nullptr, 0, nullptr, 0, nullptr);
  // dense2: out = relu(cat2 @ W2 + b2) + h
  gemm_bf16_wmma<false, 2><<<g, 256, 0, stream>>>(
      cat2, W2T, 1024, b2, nullptr, 0, 0, out, 512, nullptr, 0, hF);
}
